// TrainableESN_57054345560417
// MI455X (gfx1250) — matrix operand order repaired and checked by reading, not compiled, not run on hardware
//
#include <hip/hip_runtime.h>
#include <hip/hip_bf16.h>
#include <math.h>

// ---------------- problem constants (from reference) ----------------
#define ESN_B   32
#define ESN_U   2048
#define ESN_F   64
#define ESN_T   512
#define LEAKY_  0.1f
#define MASKV   (-2.0f)

// ---------------- kernel configuration ----------------
#define NCOLS    64                 // units (columns of W_hat) owned per workgroup
#define NWG      (ESN_U / NCOLS)    // 32 workgroups
#define NTHREADS 256                // 8 wave32 waves: 2 M-tiles x 4 N-tiles of 16x16
#define CHUNK_K  256                // K-chunk of h staged per TDM transfer
#define NCHUNK   (ESN_U / CHUNK_K)  // 8 chunks per timestep

typedef __bf16 bf16_t;
typedef __attribute__((ext_vector_type(16))) __bf16 v16bf;
typedef __attribute__((ext_vector_type(8)))  float  v8f;
typedef unsigned int v4u __attribute__((ext_vector_type(4)));
typedef int          v8i __attribute__((ext_vector_type(8)));
typedef int          v4i __attribute__((ext_vector_type(4)));

// ---------------- workspace layout (bytes) ----------------
#define WS_HB0   0                                   // h ping  : [B][U] bf16 (128KB)
#define WS_HB1   (ESN_B * ESN_U * 2)                 // h pong  : [B][U] bf16 (128KB)
#define WS_PART  (2 * ESN_B * ESN_U * 2)             // partial readout: [NWG][B] f32
#define WS_BAR   (WS_PART + NWG * ESN_B * 4)         // barrier: {count, generation}
#define WS_TOTAL (WS_BAR + 8)

// ---------------- LDS layout (bytes; dynamic LDS starts at offset 0) ----------------
#define LDS_WSL  0                                   // [NCOLS][U]  W_hat slice bf16 (256KB)
#define LDS_WIN  (NCOLS * ESN_U * 2)                 // [NCOLS][F]  W_in slice  bf16 (8KB)
#define LDS_SB   (LDS_WIN + NCOLS * ESN_F * 2)       // [NCOLS]     bias f32
#define LDS_SM   (LDS_SB + NCOLS * 4)                // [B]         step mask u32
#define LDS_STG  (LDS_SM + ESN_B * 4)                // 2x [B][CHUNK_K] bf16 h stage (2x16KB)
#define CHUNK_BYTES (ESN_B * CHUNK_K * 2)            // 16384
#define SMEM_BYTES  (LDS_STG + 2 * CHUNK_BYTES)      // 303488 <= 320KB WGP LDS

// ---------------- TDM: DMA one 32xCHUNK_K bf16 tile of h into an LDS stage buffer ----
__device__ __forceinline__ void tdm_load_chunk(const bf16_t* hsrc, int chunk, int buf) {
  unsigned long long ga = (unsigned long long)(const void*)hsrc
                        + (unsigned long long)chunk * (CHUNK_K * 2);
  v4u g0;
  g0[0] = 1u;                                           // count=1 (valid user D#)
  g0[1] = (unsigned)(LDS_STG + buf * CHUNK_BYTES);      // lds_addr
  g0[2] = (unsigned)(ga & 0xffffffffu);                 // global_addr[31:0]
  g0[3] = ((unsigned)(ga >> 32) & 0x01ffffffu)          // global_addr[56:32]
        | 0x80000000u;                                  // type=2 ("image")
  v8i g1;
  g1[0] = 0x00010000;                                   // data_size=1 (2 bytes), no multicast
  g1[1] = (int)((ESN_U & 0xffff) << 16);                // tensor_dim0 = 2048 (bits 79:48)
  g1[2] = (int)((ESN_U >> 16) | (ESN_B << 16));         // dim0 hi | tensor_dim1 = 32
  g1[3] = (int)(CHUNK_K << 16);                         // tile_dim0 = 256 (bits 127:112)
  g1[4] = ESN_B;                                        // tile_dim1 = 32, tile_dim2 = 0
  g1[5] = ESN_U;                                        // tensor_dim0_stride = 2048
  g1[6] = 0;
  g1[7] = 0;
  v4i z4 = {0, 0, 0, 0};
#if __clang_major__ >= 23
  v8i z8 = {0, 0, 0, 0, 0, 0, 0, 0};
  __builtin_amdgcn_tensor_load_to_lds(g0, g1, z4, z4, z8, 0);
#else
  __builtin_amdgcn_tensor_load_to_lds(g0, g1, z4, z4, 0);
#endif
}

// ---------------- grid-wide barrier (persistent kernel) ----------------
__device__ __forceinline__ void grid_barrier(volatile unsigned* cnt,
                                             volatile unsigned* gen) {
  __syncthreads();
  if (threadIdx.x == 0) {
    __threadfence();                              // h stores visible device-wide
    unsigned g = *gen;
    unsigned prev = atomicAdd((unsigned*)cnt, 1u);
    if (prev == (unsigned)(NWG - 1)) {
      *cnt = 0u;
      __threadfence();
      atomicAdd((unsigned*)gen, 1u);              // release everyone
    } else {
      while (*gen == g) { __builtin_amdgcn_s_sleep(2); }
    }
  }
  __threadfence();
  __syncthreads();
}

// ---------------- init: zero the workspace ----------------
__global__ void esn_init(unsigned* __restrict__ p, int n) {
  int i = blockIdx.x * blockDim.x + threadIdx.x;
  if (i < n) p[i] = 0u;
}

// ---------------- persistent ESN kernel ----------------
__global__ __launch_bounds__(NTHREADS)
void esn_persistent(const float* __restrict__ x,      // [B][T][F]
                    const float* __restrict__ Win,    // [F][U]
                    const float* __restrict__ What,   // [U][U]  (k-major: W[k][n])
                    const float* __restrict__ bias,   // [U]
                    const float* __restrict__ Wout,   // [U][1]
                    char* __restrict__ ws)
{
  extern __shared__ char smem[];
  bf16_t*   WSL = (bf16_t*)(smem + LDS_WSL);          // [NCOLS][U]  W_hat slice (bf16)
  bf16_t*   WIN = (bf16_t*)(smem + LDS_WIN);          // [NCOLS][F]  W_in slice  (bf16)
  float*    SB  = (float*)(smem + LDS_SB);            // [NCOLS]     bias slice
  unsigned* SM  = (unsigned*)(smem + LDS_SM);         // [B]         step mask
  float*    RED = (float*)(smem + LDS_STG);           // [B][NCOLS]  (aliases stage bufs)

  bf16_t*   hb0  = (bf16_t*)(ws + WS_HB0);
  bf16_t*   hb1  = (bf16_t*)(ws + WS_HB1);
  float*    part = (float*)(ws + WS_PART);
  unsigned* bcnt = (unsigned*)(ws + WS_BAR);
  unsigned* bgen = bcnt + 1;

  const int tid   = threadIdx.x;
  const int wg    = blockIdx.x;
  const int n0    = wg * NCOLS;
  const int wave  = tid >> 5;
  const int lane  = tid & 31;
  const int mtile = wave & 1;          // M block: rows 0-15 / 16-31 of batch
  const int ntile = wave >> 1;         // N block: 4 tiles of 16 columns
  const int lrow  = lane & 15;
  const int khalf = lane >> 4;         // lane half selects K sub-block per ISA layout
  const int ncol  = ntile * 16 + lrow; // 0..63 local column
  const int nglob = n0 + ncol;

  // ---- one-time: stage weight slices into LDS as bf16 (coalesced global reads) ----
  for (int i = tid; i < NCOLS * ESN_U; i += NTHREADS) {
    int nl = i % NCOLS, k = i / NCOLS;
    WSL[(size_t)nl * ESN_U + k] = (bf16_t)What[(size_t)k * ESN_U + n0 + nl];
  }
  for (int i = tid; i < NCOLS * ESN_F; i += NTHREADS) {
    int nl = i % NCOLS, f = i / NCOLS;
    WIN[nl * ESN_F + f] = (bf16_t)Win[f * ESN_U + n0 + nl];
  }
  if (tid < NCOLS) SB[tid] = bias[n0 + tid];
  __syncthreads();

  // per-thread fp32 recurrent state: element v -> (b = mtile*16 + khalf*8 + v, n = nglob)
  float hreg[8], ssum[8];
#pragma unroll
  for (int v = 0; v < 8; ++v) { hreg[v] = 0.f; ssum[v] = 0.f; }
  const float bval = SB[ncol];

  const float* xrow = x + ((size_t)(mtile * 16 + lrow) * ESN_T) * ESN_F;

  for (int t = 0; t < ESN_T; ++t) {
    const bf16_t* hc = (t & 1) ? hb1 : hb0;   // read
    bf16_t*       hw = (t & 1) ? hb0 : hb1;   // write

    // kick off DMA of the first h chunk while we do mask + input term
    if (wave == 0) tdm_load_chunk(hc, 0, 0);

    // mask[b] = any(x[b,t,:] != MASKV)
    if (tid < ESN_B) {
      const float* xr = x + ((size_t)tid * ESN_T + t) * ESN_F;
      unsigned m = 0;
#pragma unroll 8
      for (int f = 0; f < ESN_F; ++f) m |= (xr[f] != MASKV) ? 1u : 0u;
      SM[tid] = m;
    }

    v8f c = {0.f, 0.f, 0.f, 0.f, 0.f, 0.f, 0.f, 0.f};

    // ---- input term: x_t @ W_in  (K = 64 -> 2 bf16 WMMA steps, overlaps TDM) ----
    const float* xt = xrow + (size_t)t * ESN_F;
#pragma unroll
    for (int kk = 0; kk < 2; ++kk) {
      const int kb0 = kk * 32 + khalf * 8;       // A: K 0-7 / 8-15 (+16 for hi half)
      v16bf a;
#pragma unroll
      for (int i = 0; i < 8; ++i) {
        a[i]     = (bf16_t)xt[kb0 + i];
        a[8 + i] = (bf16_t)xt[kb0 + 16 + i];
      }
      const bf16_t* wr = WIN + ncol * ESN_F + kk * 32 + khalf * 8;   // B: K 0-7 / 8-15 (+16 for hi half)
      v16bf bm;
      ((uint4*)&bm)[0] = ((const uint4*)wr)[0];
      ((uint4*)&bm)[1] = ((const uint4*)(wr + 16))[0];
      c = __builtin_amdgcn_wmma_f32_16x16x32_bf16(false, a, false, bm,
                                                  (short)0, c, false, false);
    }

    // ---- recurrent term: h @ W_hat slice, TDM double-buffered K-chunks ----
    for (int ch = 0; ch < NCHUNK; ++ch) {
      if (wave == 0) {
        if (ch < NCHUNK - 1) {
          tdm_load_chunk(hc, ch + 1, (ch + 1) & 1);        // prefetch next chunk
          __builtin_amdgcn_s_wait_tensorcnt((short)1);     // current chunk landed
        } else {
          __builtin_amdgcn_s_wait_tensorcnt((short)0);     // last chunk landed
        }
      }
      __syncthreads();                                     // stage[ch&1] ready for all

      const bf16_t* stg  = (const bf16_t*)(smem + LDS_STG + (ch & 1) * CHUNK_BYTES);
      const bf16_t* arow = stg + (mtile * 16 + lrow) * CHUNK_K;
      const bf16_t* wrow = WSL + (size_t)ncol * ESN_U + ch * CHUNK_K;
#pragma unroll
      for (int kk = 0; kk < CHUNK_K / 32; ++kk) {          // 8 WMMA per chunk
        const int kb0 = kk * 32 + khalf * 8;
        v16bf a;                                           // A: 2x b128 from LDS stage
        ((uint4*)&a)[0] = ((const uint4*)(arow + kb0))[0];
        ((uint4*)&a)[1] = ((const uint4*)(arow + kb0 + 16))[0];
        const bf16_t* wr = wrow + kk * 32 + khalf * 8;
        v16bf bm;                                          // B: 2x b128 from LDS W slice
        ((uint4*)&bm)[0] = ((const uint4*)wr)[0];
        ((uint4*)&bm)[1] = ((const uint4*)(wr + 16))[0];
        c = __builtin_amdgcn_wmma_f32_16x16x32_bf16(false, a, false, bm,
                                                    (short)0, c, false, false);
      }
      __syncthreads();                                     // done reading stage[ch&1]
    }

    // ---- fp32 leaky/tanh/mask update on privately-owned state elements ----
#pragma unroll
    for (int v = 0; v < 8; ++v) {
      const int b = mtile * 16 + khalf * 8 + v;
      float pre  = c[v] + bval;
      float cand = (1.0f - LEAKY_) * hreg[v] + LEAKY_ * tanhf(pre);
      float hv   = SM[b] ? cand : hreg[v];
      hreg[v] = hv;
      ssum[v] += hv;                                   // temporal sum for mean pool
      hw[(size_t)b * ESN_U + nglob] = (bf16_t)hv;      // publish bf16 for next step
    }

    grid_barrier(bcnt, bgen);
  }

  // ---- deterministic readout partial: (mean over T) * W_out, fixed-order sums ----
  const float scale = 1.0f / (float)ESN_T;
  const float wv = Wout[nglob];
#pragma unroll
  for (int v = 0; v < 8; ++v) {
    const int b = mtile * 16 + khalf * 8 + v;
    RED[b * NCOLS + ncol] = ssum[v] * scale * wv;      // each slot written exactly once
  }
  __syncthreads();
  if (tid < ESN_B) {
    float s = 0.f;
    for (int nc = 0; nc < NCOLS; ++nc) s += RED[tid * NCOLS + nc];
    part[wg * ESN_B + tid] = s;
  }
}

// ---------------- finalize: sum partials (fixed order) + sigmoid ----------------
__global__ void esn_finalize(const char* __restrict__ ws,
                             const float* __restrict__ bout,
                             float* __restrict__ out) {
  int b = threadIdx.x;
  if (b >= ESN_B) return;
  const float* part = (const float*)(ws + WS_PART);
  float s = bout[0];
  for (int w = 0; w < NWG; ++w) s += part[w * ESN_B + b];
  out[b] = 1.0f / (1.0f + expf(-s));
}

// ---------------- launch ----------------
extern "C" void kernel_launch(void* const* d_in, const int* in_sizes, int n_in,
                              void* d_out, int out_size, void* d_ws, size_t ws_size,
                              hipStream_t stream) {
  (void)in_sizes; (void)n_in; (void)out_size; (void)ws_size;
  const float* x    = (const float*)d_in[0];
  const float* Win  = (const float*)d_in[1];
  const float* What = (const float*)d_in[2];
  const float* bias = (const float*)d_in[3];
  const float* Wout = (const float*)d_in[4];
  const float* bout = (const float*)d_in[5];
  float* out = (float*)d_out;
  char*  ws  = (char*)d_ws;

  const int nz = WS_TOTAL / 4;
  esn_init<<<(nz + 255) / 256, 256, 0, stream>>>((unsigned*)ws, nz);
  esn_persistent<<<NWG, NTHREADS, SMEM_BYTES, stream>>>(x, Win, What, bias, Wout, ws);
  esn_finalize<<<1, ESN_B, 0, stream>>>(ws, bout, out);
}
